// TextGuidedGCN_10015863734394
// MI455X (gfx1250) — matrix operand order repaired and checked by reading, not compiled, not run on hardware
//
#include <hip/hip_runtime.h>
#include <hip/hip_bf16.h>

// ---------------------------------------------------------------------------
// Problem constants (from reference)
// ---------------------------------------------------------------------------
#define BB   8
#define NN   4096
#define DP   1024
#define HH   256
#define KS   512     // k_sample
#define KNN  8
#define NCLS 4

typedef __bf16 bf16_t;
typedef bf16_t v16bf __attribute__((ext_vector_type(16)));
typedef bf16_t v8bf  __attribute__((ext_vector_type(8)));
typedef float  v8f   __attribute__((ext_vector_type(8)));

__device__ __forceinline__ bf16_t f2bf(float f) {
  unsigned int u = __builtin_bit_cast(unsigned int, f);
  unsigned int r = u + 0x7FFFu + ((u >> 16) & 1u);   // round-to-nearest-even
  unsigned short h = (unsigned short)(r >> 16);
  return __builtin_bit_cast(bf16_t, h);
}

// ---------------------------------------------------------------------------
// Weight prep: f32 [K,N] -> bf16 transposed [N,K]
// ---------------------------------------------------------------------------
__global__ void cvt_transpose_kernel(const float* __restrict__ src,
                                     bf16_t* __restrict__ dst, int K, int N) {
  const int gid = blockIdx.x * blockDim.x + threadIdx.x;
  if (gid < K * N) {
    const int k = gid / N;
    const int n = gid - k * N;
    dst[(size_t)n * K + k] = f2bf(src[gid]);
  }
}

// ---------------------------------------------------------------------------
// WMMA bf16 GEMM, N fixed at 256:  C = act(A[M,K] @ W[K,256] + bias)
//   A: f32 row-major, converted to bf16 while staging into LDS (read ONCE).
//   Wt: bf16 [256][K] (transposed) -> B fragments loaded straight from
//       global as contiguous 32B chunks (weights are L2-resident).
//   Block: 256 threads / 8 waves; tile M=64 x N=256, K-step 32.
//   Wave w owns cols [32w, 32w+32) : 2 N-subtiles x 4 M-subtiles = 8 WMMA/step.
//   scoreMode==1: instead of storing C, fuse the text-similarity scoring:
//       score[row] = dot(v_row, txt) / ||v_row||   (v = A@W + bias)
//   M % 64 == 0, K % 32 == 0. EXEC stays all-ones everywhere (WMMA req).
// ---------------------------------------------------------------------------
__global__ __launch_bounds__(256)
void gemm_wmma_n256_kernel(const float* __restrict__ A,
                           const bf16_t* __restrict__ Wt,
                           const float* __restrict__ bias,
                           float* __restrict__ C,
                           int M, int K, int act,
                           const float* __restrict__ txtEmb,
                           float* __restrict__ scores, int scoreMode) {
  __shared__ __align__(64) bf16_t ldsA[64 * 32];   // [m][k]
  __shared__ float sDot[64];
  __shared__ float sSS[64];

  const int tid  = threadIdx.x;
  const int lane = tid & 31;
  const int wave = tid >> 5;
  const int blockM = blockIdx.x * 64;
  const int nb = wave * 32;              // this wave's column slab

  if (tid < 64) { sDot[tid] = 0.0f; sSS[tid] = 0.0f; }

  v8f acc[8];
#pragma unroll
  for (int a = 0; a < 8; ++a) acc[a] = (v8f){};

  // A staging: 8 contiguous floats per thread
  const int ar = tid >> 2;           // 0..63 row in tile
  const int ak = (tid & 3) * 8;      // 0,8,16,24

  const int laneM = lane & 15;
  const int kbA   = (lane >> 4) * 16;    // 0 or 16  (A fragment K base)
  const int kbB   = (lane >> 4) * 16;    // 0 or 16  (B fragment K base)

  const bf16_t* wt0 = Wt + (size_t)(nb + laneM) * K;
  const bf16_t* wt1 = Wt + (size_t)(nb + 16 + laneM) * K;

  for (int k0 = 0; k0 < K; k0 += 32) {
    // ---- stage A tile (f32 -> bf16 into LDS) ----
    const float* Ap = A + (size_t)(blockM + ar) * K + k0 + ak;
    float4 f0 = *(const float4*)(Ap);
    float4 f1 = *(const float4*)(Ap + 4);
    if (k0 + 32 < K) __builtin_prefetch(Ap + 32, 0, 0);
    bf16_t* da = &ldsA[ar * 32 + ak];
    da[0] = f2bf(f0.x); da[1] = f2bf(f0.y); da[2] = f2bf(f0.z); da[3] = f2bf(f0.w);
    da[4] = f2bf(f1.x); da[5] = f2bf(f1.y); da[6] = f2bf(f1.z); da[7] = f2bf(f1.w);

    // ---- B fragments straight from global (L2-resident weights) ----
    v16bf bf0 = *(const v16bf*)(wt0 + k0 + kbB);
    v16bf bf1 = *(const v16bf*)(wt1 + k0 + kbB);

    __syncthreads();

    // ---- 4 M-subtiles x 2 N-subtiles -> 8 WMMA per K-step ----
#pragma unroll
    for (int m = 0; m < 4; ++m) {
      const int row = m * 16 + laneM;
      v8bf alo = *(const v8bf*)&ldsA[row * 32 + kbA];
      v8bf ahi = *(const v8bf*)&ldsA[row * 32 + 8 + kbA];
      v16bf af = __builtin_shufflevector(alo, ahi,
          0,1,2,3,4,5,6,7,8,9,10,11,12,13,14,15);
      acc[m * 2 + 0] = __builtin_amdgcn_wmma_f32_16x16x32_bf16(
          false, af, false, bf0, (short)0, acc[m * 2 + 0], false, false);
      acc[m * 2 + 1] = __builtin_amdgcn_wmma_f32_16x16x32_bf16(
          false, af, false, bf1, (short)0, acc[m * 2 + 1], false, false);
    }
    __syncthreads();
  }

  // ---- epilogue ----
  const int rbase = (lane >> 4) * 8;   // C layout: VGPR r -> M=r (+8 hi lanes)
  const int col0 = nb + laneM;
  const int col1 = nb + 16 + laneM;
  const float bv0 = bias[col0];
  const float bv1 = bias[col1];

  if (scoreMode) {
    const int b = blockM >> 12;        // 4096 rows per batch
    const float t0 = txtEmb[b * HH + col0];
    const float t1 = txtEmb[b * HH + col1];
#pragma unroll
    for (int m = 0; m < 4; ++m) {
#pragma unroll
      for (int r = 0; r < 8; ++r) {
        const float v0 = acc[m * 2 + 0][r] + bv0;
        const float v1 = acc[m * 2 + 1][r] + bv1;
        float d = v0 * t0 + v1 * t1;
        float s = v0 * v0 + v1 * v1;
#pragma unroll
        for (int o = 1; o < 16; o <<= 1) {   // reduce within 16-lane row group
          d += __shfl_xor(d, o, 32);
          s += __shfl_xor(s, o, 32);
        }
        if (laneM == 0) {
          const int rowl = m * 16 + rbase + r;
          atomicAdd(&sDot[rowl], d);
          atomicAdd(&sSS[rowl], s);
        }
      }
    }
    __syncthreads();
    if (tid < 64)
      scores[blockM + tid] = sDot[tid] / fmaxf(sqrtf(sSS[tid]), 1e-12f);
  } else {
#pragma unroll
    for (int m = 0; m < 4; ++m) {
#pragma unroll
      for (int r = 0; r < 8; ++r) {
        float v0 = acc[m * 2 + 0][r] + bv0;
        float v1 = acc[m * 2 + 1][r] + bv1;
        if (act == 1) { v0 = fmaxf(v0, 0.0f); v1 = fmaxf(v1, 0.0f); }
        else if (act == 2) { v0 = tanhf(v0); v1 = tanhf(v1); }
        else if (act == 3) { v0 = 1.0f / (1.0f + expf(-v0));
                             v1 = 1.0f / (1.0f + expf(-v1)); }
        const int row = blockM + m * 16 + rbase + r;
        C[(size_t)row * HH + col0] = v0;
        C[(size_t)row * HH + col1] = v1;
      }
    }
  }
}

// ---------------------------------------------------------------------------
// Text branch: txt_emb[b] = l2norm(text[b] @ W_txt + b_txt)
// ---------------------------------------------------------------------------
__global__ __launch_bounds__(256)
void txt_kernel(const float* __restrict__ text, const float* __restrict__ Wt,
                const float* __restrict__ bt, float* __restrict__ txtEmb) {
  __shared__ float emb[HH];
  __shared__ float red[HH];
  const int b = blockIdx.x, h = threadIdx.x;
  float acc = bt[h];
  const float* tr = text + (size_t)b * DP;
  for (int d = 0; d < DP; ++d) acc += tr[d] * Wt[(size_t)d * HH + h];
  emb[h] = acc;
  red[h] = acc * acc;
  __syncthreads();
  for (int s = HH / 2; s > 0; s >>= 1) {
    if (h < s) red[h] += red[h + s];
    __syncthreads();
  }
  const float nrm = fmaxf(sqrtf(red[0]), 1e-12f);
  txtEmb[b * HH + h] = emb[h] / nrm;
}

// ---------------------------------------------------------------------------
// Per-batch descending bitonic sort of 4096 (score, idx); emit top 512 indices
// ---------------------------------------------------------------------------
__global__ __launch_bounds__(1024)
void topk_kernel(const float* __restrict__ scores, int* __restrict__ idx) {
  __shared__ float sv[NN];
  __shared__ int   si[NN];
  const int b = blockIdx.x;
  for (int i = threadIdx.x; i < NN; i += 1024) {
    sv[i] = scores[b * NN + i];
    si[i] = i;
  }
  __syncthreads();
  for (int size = 2; size <= NN; size <<= 1) {
    for (int stride = size >> 1; stride > 0; stride >>= 1) {
      for (int t = threadIdx.x; t < NN / 2; t += 1024) {
        const int i = ((t & ~(stride - 1)) << 1) | (t & (stride - 1));
        const int p = i + stride;
        const bool up = ((i & size) == 0);
        if ((sv[i] < sv[p]) == up) {      // descending overall
          const float tv = sv[i]; sv[i] = sv[p]; sv[p] = tv;
          const int   ti = si[i]; si[i] = si[p]; si[p] = ti;
        }
      }
      __syncthreads();
    }
  }
  for (int j = threadIdx.x; j < KS; j += 1024) idx[b * KS + j] = si[j];
}

// ---------------------------------------------------------------------------
// Gather selected patches (features + coords)
// ---------------------------------------------------------------------------
__global__ void gather_kernel(const float* __restrict__ x,
                              const float* __restrict__ coords,
                              const int* __restrict__ idx,
                              float* __restrict__ selX,
                              float* __restrict__ selPos) {
  const int gid = blockIdx.x * blockDim.x + threadIdx.x;   // B*KS*DP
  const int b = gid >> 19;
  const int rem = gid & ((1 << 19) - 1);
  const int i = rem >> 10;
  const int f = rem & (DP - 1);
  const int src = idx[b * KS + i];
  selX[gid] = x[((size_t)b * NN + src) * DP + f];
  if (f < 2) selPos[(b * KS + i) * 2 + f] = coords[((size_t)b * NN + src) * 2 + f];
}

// ---------------------------------------------------------------------------
// kNN-8 (exclude self) on selected coordinates; block = one batch
// ---------------------------------------------------------------------------
__global__ __launch_bounds__(512)
void knn_kernel(const float* __restrict__ selPos, int* __restrict__ nbr) {
  __shared__ float px[KS];
  __shared__ float py[KS];
  const int b = blockIdx.x, i = threadIdx.x;
  px[i] = selPos[(b * KS + i) * 2 + 0];
  py[i] = selPos[(b * KS + i) * 2 + 1];
  __syncthreads();
  float bd[KNN];
  int   bi[KNN];
#pragma unroll
  for (int t = 0; t < KNN; ++t) { bd[t] = 3.0e38f; bi[t] = 0; }
  const float xi = px[i], yi = py[i];
  for (int j = 0; j < KS; ++j) {
    if (j == i) continue;
    const float dx = px[j] - xi, dy = py[j] - yi;
    const float d = dx * dx + dy * dy;
    if (d < bd[KNN - 1]) {
      int t = KNN - 1;
      while (t > 0 && bd[t - 1] > d) { bd[t] = bd[t - 1]; bi[t] = bi[t - 1]; --t; }
      bd[t] = d; bi[t] = j;
    }
  }
#pragma unroll
  for (int t = 0; t < KNN; ++t) nbr[(b * KS + i) * KNN + t] = bi[t];
}

// ---------------------------------------------------------------------------
// GIN aggregation: z[b,i,:] = h[b,i,:] + sum_t h[b, nbr[b,i,t], :]
// ---------------------------------------------------------------------------
__global__ void aggregate_kernel(const float* __restrict__ h,
                                 const int* __restrict__ nbr,
                                 float* __restrict__ z, int Hdim) {
  const int gid = blockIdx.x * blockDim.x + threadIdx.x;   // B*KS*Hdim
  const int f = gid % Hdim;
  const int node = gid / Hdim;                             // b*KS + i
  const int* nb = nbr + (size_t)node * KNN;
  const int b = node / KS;
  float s = h[(size_t)node * Hdim + f];
#pragma unroll
  for (int t = 0; t < KNN; ++t)
    s += h[((size_t)b * KS + nb[t]) * Hdim + f];
  z[gid] = s;
}

// ---------------------------------------------------------------------------
// Attention logits: logit = sum_h tanh(.)*sigm(.) * Wc[h] + bc; wave per row
// ---------------------------------------------------------------------------
__global__ __launch_bounds__(256)
void attn_logit_kernel(const float* __restrict__ Aa, const float* __restrict__ Ag,
                       const float* __restrict__ Wc, const float* __restrict__ bc,
                       float* __restrict__ logits) {
  const int row  = blockIdx.x * 8 + (threadIdx.x >> 5);
  const int lane = threadIdx.x & 31;
  const float* pa = Aa + (size_t)row * HH;
  const float* pg = Ag + (size_t)row * HH;
  float acc = 0.0f;
  for (int j = lane; j < HH; j += 32) acc += pa[j] * pg[j] * Wc[j];
  for (int o = 16; o > 0; o >>= 1) acc += __shfl_xor(acc, o, 32);
  if (lane == 0) logits[row] = acc + bc[0];
}

// ---------------------------------------------------------------------------
// Per-batch softmax over 512 logits
// ---------------------------------------------------------------------------
__global__ __launch_bounds__(512)
void softmax_kernel(const float* __restrict__ logits, float* __restrict__ w) {
  __shared__ float red[KS];
  const int b = blockIdx.x, i = threadIdx.x;
  const float v = logits[b * KS + i];
  red[i] = v;
  __syncthreads();
  for (int s = KS / 2; s > 0; s >>= 1) {
    if (i < s) red[i] = fmaxf(red[i], red[i + s]);
    __syncthreads();
  }
  const float m = red[0];
  __syncthreads();
  const float e = expf(v - m);
  red[i] = e;
  __syncthreads();
  for (int s = KS / 2; s > 0; s >>= 1) {
    if (i < s) red[i] += red[i + s];
    __syncthreads();
  }
  w[b * KS + i] = e / red[0];
}

// ---------------------------------------------------------------------------
// pooled[b,h] = sum_k w[b,k] * h3[b,k,h]
// ---------------------------------------------------------------------------
__global__ __launch_bounds__(256)
void pool_kernel(const float* __restrict__ w, const float* __restrict__ h3,
                 float* __restrict__ pooled) {
  const int b = blockIdx.x, h = threadIdx.x;
  float acc = 0.0f;
  for (int k = 0; k < KS; ++k)
    acc += w[b * KS + k] * h3[((size_t)b * KS + k) * HH + h];
  pooled[b * HH + h] = acc;
}

// ---------------------------------------------------------------------------
// head: out = relu(pooled @ Wr + br) @ Wcls + bcls
// ---------------------------------------------------------------------------
__global__ __launch_bounds__(256)
void head_kernel(const float* __restrict__ pooled, const float* __restrict__ Wr,
                 const float* __restrict__ br, const float* __restrict__ Wcls,
                 const float* __restrict__ bcls, float* __restrict__ out) {
  __shared__ float hr[HH];
  const int b = blockIdx.x, h = threadIdx.x;
  float acc = br[h];
  for (int d = 0; d < HH; ++d) acc += pooled[b * HH + d] * Wr[d * HH + h];
  hr[h] = fmaxf(acc, 0.0f);
  __syncthreads();
  if (h < NCLS) {
    float o = bcls[h];
    for (int d = 0; d < HH; ++d) o += hr[d] * Wcls[d * NCLS + h];
    out[b * NCLS + h] = o;
  }
}

// ---------------------------------------------------------------------------
// Orchestration
// ---------------------------------------------------------------------------
static inline size_t align_up(size_t x, size_t a) { return (x + a - 1) / a * a; }

extern "C" void kernel_launch(void* const* d_in, const int* in_sizes, int n_in,
                              void* d_out, int out_size, void* d_ws, size_t ws_size,
                              hipStream_t stream) {
  const float* x      = (const float*)d_in[0];
  const float* coords = (const float*)d_in[1];
  const float* text   = (const float*)d_in[2];
  const float* W_img  = (const float*)d_in[3];
  const float* b_img  = (const float*)d_in[4];
  const float* W_txt  = (const float*)d_in[5];
  const float* b_txt  = (const float*)d_in[6];
  const float* W1a = (const float*)d_in[7],  *b1a = (const float*)d_in[8];
  const float* W1b = (const float*)d_in[9],  *b1b = (const float*)d_in[10];
  const float* W2a = (const float*)d_in[11], *b2a = (const float*)d_in[12];
  const float* W2b = (const float*)d_in[13], *b2b = (const float*)d_in[14];
  const float* W3a = (const float*)d_in[15], *b3a = (const float*)d_in[16];
  const float* W3b = (const float*)d_in[17], *b3b = (const float*)d_in[18];
  const float* Wa  = (const float*)d_in[19], *ba  = (const float*)d_in[20];
  const float* Wg  = (const float*)d_in[21], *bg  = (const float*)d_in[22];
  const float* Wc  = (const float*)d_in[23], *bc  = (const float*)d_in[24];
  const float* Wr  = (const float*)d_in[25], *br  = (const float*)d_in[26];
  const float* Wcls = (const float*)d_in[27], *bcls = (const float*)d_in[28];

  // ---- workspace carve ----
  char* p = (char*)d_ws;
  auto alloc = [&](size_t bytes) -> void* {
    void* r = (void*)p;
    p += align_up(bytes, 256);
    return r;
  };
  bf16_t* wtImg = (bf16_t*)alloc((size_t)DP * HH * 2);   // transposed [N][K]
  bf16_t* wt1a  = (bf16_t*)alloc((size_t)DP * HH * 2);
  bf16_t* wt1b  = (bf16_t*)alloc((size_t)HH * HH * 2);
  bf16_t* wt2a  = (bf16_t*)alloc((size_t)HH * HH * 2);
  bf16_t* wt2b  = (bf16_t*)alloc((size_t)HH * HH * 2);
  bf16_t* wt3a  = (bf16_t*)alloc((size_t)HH * HH * 2);
  bf16_t* wt3b  = (bf16_t*)alloc((size_t)HH * HH * 2);
  bf16_t* wtA   = (bf16_t*)alloc((size_t)HH * HH * 2);
  bf16_t* wtG   = (bf16_t*)alloc((size_t)HH * HH * 2);
  float* txtEmb = (float*)alloc((size_t)BB * HH * 4);
  float* scores = (float*)alloc((size_t)BB * NN * 4);
  int*   idxBuf = (int*)alloc((size_t)BB * KS * 4);
  float* selX   = (float*)alloc((size_t)BB * KS * DP * 4);
  float* selPos = (float*)alloc((size_t)BB * KS * 2 * 4);
  int*   nbr    = (int*)alloc((size_t)BB * KS * KNN * 4);
  float* zBig   = (float*)alloc((size_t)BB * KS * DP * 4);
  float* tBuf   = (float*)alloc((size_t)BB * KS * HH * 4);
  float* h1     = (float*)alloc((size_t)BB * KS * HH * 4);
  float* zS     = (float*)alloc((size_t)BB * KS * HH * 4);
  float* h2     = (float*)alloc((size_t)BB * KS * HH * 4);
  float* h3     = (float*)alloc((size_t)BB * KS * HH * 4);
  float* AaB    = (float*)alloc((size_t)BB * KS * HH * 4);
  float* AgB    = (float*)alloc((size_t)BB * KS * HH * 4);
  float* logits = (float*)alloc((size_t)BB * KS * 4);
  float* attnW  = (float*)alloc((size_t)BB * KS * 4);
  float* pooled = (float*)alloc((size_t)BB * HH * 4);

  // ---- weight prep (f32 [K,N] -> bf16 [N][K]) ----
  const int CT = 256;
  cvt_transpose_kernel<<<(DP * HH + CT - 1) / CT, CT, 0, stream>>>(W_img, wtImg, DP, HH);
  cvt_transpose_kernel<<<(DP * HH + CT - 1) / CT, CT, 0, stream>>>(W1a, wt1a, DP, HH);
  cvt_transpose_kernel<<<(HH * HH + CT - 1) / CT, CT, 0, stream>>>(W1b, wt1b, HH, HH);
  cvt_transpose_kernel<<<(HH * HH + CT - 1) / CT, CT, 0, stream>>>(W2a, wt2a, HH, HH);
  cvt_transpose_kernel<<<(HH * HH + CT - 1) / CT, CT, 0, stream>>>(W2b, wt2b, HH, HH);
  cvt_transpose_kernel<<<(HH * HH + CT - 1) / CT, CT, 0, stream>>>(W3a, wt3a, HH, HH);
  cvt_transpose_kernel<<<(HH * HH + CT - 1) / CT, CT, 0, stream>>>(W3b, wt3b, HH, HH);
  cvt_transpose_kernel<<<(HH * HH + CT - 1) / CT, CT, 0, stream>>>(Wa, wtA, HH, HH);
  cvt_transpose_kernel<<<(HH * HH + CT - 1) / CT, CT, 0, stream>>>(Wg, wtG, HH, HH);

  const int Mbig = BB * NN;   // 32768
  const int Msel = BB * KS;   // 4096

  // ---- text embedding first (needed by fused scoring epilogue) ----
  txt_kernel<<<BB, HH, 0, stream>>>(text, W_txt, b_txt, txtEmb);

  // ---- stage 1: fused  scores = dot(l2n(x@W_img+b), txt)  (A read once) ----
  gemm_wmma_n256_kernel<<<Mbig / 64, 256, 0, stream>>>(
      x, wtImg, b_img, nullptr, Mbig, DP, 0, txtEmb, scores, 1);

  // ---- top-k, gather, kNN ----
  topk_kernel<<<BB, 1024, 0, stream>>>(scores, idxBuf);
  gather_kernel<<<(BB * KS * DP) / 256, 256, 0, stream>>>(x, coords, idxBuf, selX, selPos);
  knn_kernel<<<BB, KS, 0, stream>>>(selPos, nbr);

  // ---- GIN layer 1 ----
  aggregate_kernel<<<(BB * KS * DP) / 256, 256, 0, stream>>>(selX, nbr, zBig, DP);
  gemm_wmma_n256_kernel<<<Msel / 64, 256, 0, stream>>>(
      zBig, wt1a, b1a, tBuf, Msel, DP, 1, nullptr, nullptr, 0);
  gemm_wmma_n256_kernel<<<Msel / 64, 256, 0, stream>>>(
      tBuf, wt1b, b1b, h1, Msel, HH, 1, nullptr, nullptr, 0);

  // ---- GIN layer 2 ----
  aggregate_kernel<<<(BB * KS * HH) / 256, 256, 0, stream>>>(h1, nbr, zS, HH);
  gemm_wmma_n256_kernel<<<Msel / 64, 256, 0, stream>>>(
      zS, wt2a, b2a, tBuf, Msel, HH, 1, nullptr, nullptr, 0);
  gemm_wmma_n256_kernel<<<Msel / 64, 256, 0, stream>>>(
      tBuf, wt2b, b2b, h2, Msel, HH, 1, nullptr, nullptr, 0);

  // ---- GIN layer 3 ----
  aggregate_kernel<<<(BB * KS * HH) / 256, 256, 0, stream>>>(h2, nbr, zS, HH);
  gemm_wmma_n256_kernel<<<Msel / 64, 256, 0, stream>>>(
      zS, wt3a, b3a, tBuf, Msel, HH, 1, nullptr, nullptr, 0);
  gemm_wmma_n256_kernel<<<Msel / 64, 256, 0, stream>>>(
      tBuf, wt3b, b3b, h3, Msel, HH, 1, nullptr, nullptr, 0);

  // ---- gated attention pooling ----
  gemm_wmma_n256_kernel<<<Msel / 64, 256, 0, stream>>>(
      h3, wtA, ba, AaB, Msel, HH, 2, nullptr, nullptr, 0);   // tanh
  gemm_wmma_n256_kernel<<<Msel / 64, 256, 0, stream>>>(
      h3, wtG, bg, AgB, Msel, HH, 3, nullptr, nullptr, 0);   // sigmoid
  attn_logit_kernel<<<Msel / 8, 256, 0, stream>>>(AaB, AgB, Wc, bc, logits);
  softmax_kernel<<<BB, KS, 0, stream>>>(logits, attnW);
  pool_kernel<<<BB, HH, 0, stream>>>(attnW, h3, pooled);

  // ---- head ----
  head_kernel<<<BB, HH, 0, stream>>>(pooled, Wr, br, Wcls, bcls, (float*)d_out);

  (void)in_sizes; (void)n_in; (void)out_size; (void)ws_size;
}
